// Summary_Maker_18305150616323
// MI455X (gfx1250) — compile-verified, matrix operand order checked by reading, not run on hardware
//
#include <hip/hip_runtime.h>
#include <stdint.h>
#include <stddef.h>

// Problem constants (from reference)
#define NN 262144
#define DD 256
#define BB 1024
#define PER 256
#define NITERS 3

typedef __attribute__((ext_vector_type(16))) __bf16 v16bf;
typedef __attribute__((ext_vector_type(8)))  float  v8f;
typedef __attribute__((ext_vector_type(4)))  unsigned int uint4v;

union FragU { v16bf v; uint4v u[2]; };

__device__ __forceinline__ float bf2f(unsigned short u) {
  unsigned int x = ((unsigned int)u) << 16;
  float f; __builtin_memcpy(&f, &x, 4); return f;
}
__device__ __forceinline__ unsigned short f2bf(float f) {
  unsigned int x; __builtin_memcpy(&x, &f, 4);
  unsigned int r = x + 0x7fffu + ((x >> 16) & 1u);
  return (unsigned short)(r >> 16);
}
__device__ __forceinline__ float sigmoidf(float x) { return 1.f / (1.f + __expf(-x)); }

__device__ __forceinline__ v8f vzero8() {
  v8f z;
#pragma unroll
  for (int i = 0; i < 8; ++i) z[i] = 0.f;
  return z;
}

__device__ __forceinline__ v8f wmma_bf16(v16bf a, v16bf b, v8f c) {
  // D = A(16x32 bf16) * B(32x16 bf16) + C(f32)
  return __builtin_amdgcn_wmma_f32_16x16x32_bf16(false, a, false, b, (short)0, c, false, false);
}

// CDNA5 async copy: global -> LDS, 16B per lane, tracked by ASYNCcnt.
// lds_byte_off = low 32 bits of a generic LDS pointer (flat aperture rule).
__device__ __forceinline__ void async_ld_b128(unsigned int lds_byte_off, const void* gsrc) {
  asm volatile("global_load_async_to_lds_b128 %0, %1, off"
               :: "v"(lds_byte_off), "v"(gsrc)
               : "memory");
}
__device__ __forceinline__ void wait_asynccnt0() {
  asm volatile("s_wait_asynccnt 0" ::: "memory");
}
__device__ __forceinline__ unsigned int lds_off32(const void* p) {
  return (unsigned int)(uintptr_t)p;
}

// Load a 16x32 bf16 fragment (A: M-major rows, or B: N-major rows of W[out][in]).
// ISA layout: lanes 0-15 hold K {kb..kb+7, kb+16..kb+23}; lanes 16-31 hold {+8}.
__device__ __forceinline__ v16bf load_frag(const unsigned short* base, int ld,
                                           int major, int kbase, int lane) {
  int m   = major + (lane & 15);
  int klo = kbase + ((lane >> 4) << 3);
  const unsigned short* p = base + m * ld + klo;
  FragU f;
  f.u[0] = *(const uint4v*)(p);
  f.u[1] = *(const uint4v*)(p + 16);
  return f.v;
}

// One wave: C[16 x (8*16)] += A[16 x 256] * B[nb.. , 256]^T, A/B bf16 row-major-K.
__device__ __forceinline__ void gemm8(const unsigned short* A, int mb,
                                      const unsigned short* Bm, int nb,
                                      v8f acc[8], int lane) {
#pragma unroll
  for (int t = 0; t < 8; ++t) acc[t] = vzero8();
  for (int kk = 0; kk < 256; kk += 32) {
    v16bf a = load_frag(A, 264, mb, kk, lane);
#pragma unroll
    for (int t = 0; t < 8; ++t) {
      v16bf b = load_frag(Bm, 256, nb + t * 16, kk, lane);
      acc[t] = wmma_bf16(a, b, acc[t]);
    }
  }
}

// LayerNorm one row of 256 f32: wave-cooperative, writes bf16 (and optional f32).
__device__ __forceinline__ void ln_row(const float* src, const float* g, const float* b,
                                       unsigned short* dst_bf, float* dst_f, int lane) {
  float x[8];
  const float4* s4 = (const float4*)(src + lane * 8);
  float4 a0 = s4[0], a1 = s4[1];
  x[0] = a0.x; x[1] = a0.y; x[2] = a0.z; x[3] = a0.w;
  x[4] = a1.x; x[5] = a1.y; x[6] = a1.z; x[7] = a1.w;
  float sm = 0.f, sq = 0.f;
#pragma unroll
  for (int j = 0; j < 8; ++j) { sm += x[j]; sq += x[j] * x[j]; }
#pragma unroll
  for (int off = 16; off >= 1; off >>= 1) {
    sm += __shfl_xor(sm, off);
    sq += __shfl_xor(sq, off);
  }
  float mean = sm * (1.f / 256.f);
  float var  = sq * (1.f / 256.f) - mean * mean;
  float inv  = rsqrtf(var + 1e-5f);
#pragma unroll
  for (int j = 0; j < 8; ++j) {
    int c = lane * 8 + j;
    float y = (x[j] - mean) * inv * g[c] + b[c];
    dst_bf[c] = f2bf(y);
    if (dst_f) dst_f[c] = y;
  }
}

// ---------------- one-time prep kernels ----------------

// Convert all weight matrices to bf16 in workspace; gW1 split into k/q halves.
__global__ void __launch_bounds__(256) k_conv_weights(
    const float* __restrict__ Wq, const float* __restrict__ Wk, const float* __restrict__ Wv,
    const float* __restrict__ gW1, const float* __restrict__ W_ih, const float* __restrict__ W_hh,
    const float* __restrict__ mW1, const float* __restrict__ mW2,
    unsigned short* __restrict__ wbf) {
  int i = blockIdx.x * blockDim.x + threadIdx.x;
  if (i >= 851968) return;
  float v;
  if (i < 65536)        v = Wq[i];
  else if (i < 131072)  v = Wk[i - 65536];
  else if (i < 196608)  v = Wv[i - 131072];
  else if (i < 262144) { int j = i - 196608; v = gW1[(j >> 8) * 512 + (j & 255)]; }
  else if (i < 327680) { int j = i - 262144; v = gW1[(j >> 8) * 512 + 256 + (j & 255)]; }
  else if (i < 524288)  v = W_ih[i - 327680];
  else if (i < 720896)  v = W_hh[i - 524288];
  else if (i < 786432)  v = mW1[i - 720896];
  else                  v = mW2[i - 786432];
  wbf[i] = f2bf(v);
}

__global__ void __launch_bounds__(256) k_init_slots(
    const float* __restrict__ mu, const float* __restrict__ logsig,
    const float* __restrict__ noise, float* __restrict__ slots) {
  int i = blockIdx.x * blockDim.x + threadIdx.x;
  if (i >= BB * DD) return;
  int d = i & 255;
  slots[i] = mu[d] + __expf(logsig[d]) * noise[i];
}

// ---------------- LN(x) -> k (bf16), v (f32) ----------------
// 256 threads (8 waves) per block; 64 rows per block. Waves 0-3: k, 4-7: v.
__global__ void __launch_bounds__(256) k_ln_kv(
    const float* __restrict__ x, const float* __restrict__ lng, const float* __restrict__ lnb,
    const unsigned short* __restrict__ wbf,
    const float* __restrict__ bk, const float* __restrict__ bv,
    unsigned short* __restrict__ kbf, float* __restrict__ vout) {
  extern __shared__ __align__(16) unsigned char smem[];
  unsigned short* A = (unsigned short*)smem;  // [64][264] bf16 LN(x)
  const int lane = threadIdx.x & 31, w = threadIdx.x >> 5;
  const int rowBase = blockIdx.x * 64;
  for (int j = 0; j < 8; ++j) {
    int r = w * 8 + j;
    ln_row(x + (size_t)(rowBase + r) * DD, lng, lnb, A + r * 264, nullptr, lane);
  }
  __syncthreads();
  const bool isV = (w >= 4);
  const int strip = (w & 3);
  const unsigned short* Bmat = wbf + (isV ? 131072 : 65536);  // Wv : Wk (bf16, [out][in])
  v8f acc[16];
#pragma unroll
  for (int t = 0; t < 16; ++t) acc[t] = vzero8();
  for (int kk = 0; kk < 256; kk += 32) {
    v16bf a = load_frag(A, 264, strip * 16, kk, lane);
#pragma unroll
    for (int t = 0; t < 16; ++t) {
      v16bf b = load_frag(Bmat, 256, t * 16, kk, lane);
      acc[t] = wmma_bf16(a, b, acc[t]);
    }
  }
  const float* bias = isV ? bv : bk;
  const int hi8 = (lane >> 4) << 3;
#pragma unroll
  for (int t = 0; t < 16; ++t) {
    int col = t * 16 + (lane & 15);
    float bb = bias[col];
#pragma unroll
    for (int i = 0; i < 8; ++i) {
      int r = strip * 16 + i + hi8;
      size_t gi = (size_t)(rowBase + r) * DD + col;
      float vv = acc[t][i] + bb;
      if (isV) vout[gi] = vv;
      else     kbf[gi]  = f2bf(vv);
    }
  }
}

// ---------------- slot side: LN(slots) -> q -> qc ----------------
template <bool WRITE_SN>
__global__ void __launch_bounds__(256) k_slot_q(
    const float* __restrict__ slots, const unsigned short* __restrict__ wbf,
    const float* __restrict__ bq, const float* __restrict__ gb1,
    const float* __restrict__ lng, const float* __restrict__ lnb,
    float* __restrict__ qc, float* __restrict__ snOut) {
  extern __shared__ __align__(16) unsigned char smem[];
  unsigned short* A  = (unsigned short*)smem;   // [64][264] LN(slots) bf16
  unsigned short* A2 = A + 64 * 264;            // [64][264] q bf16
  const int lane = threadIdx.x & 31, w = threadIdx.x >> 5;
  const int rowBase = blockIdx.x * 64;
  for (int j = 0; j < 8; ++j) {
    int r = w * 8 + j;
    ln_row(slots + (size_t)(rowBase + r) * DD, lng, lnb, A + r * 264,
           WRITE_SN ? (snOut + (size_t)(rowBase + r) * DD) : nullptr, lane);
  }
  __syncthreads();
  const int strip = w & 3, nh = w >> 2;
  const int lanelo = lane & 15, hi8 = (lane >> 4) << 3;
  const int mb = strip * 16, nb = nh * 128;
  v8f acc[8];
  // q = LN(slots) @ Wq^T + bq
  gemm8(A, mb, wbf /*Wq*/, nb, acc, lane);
#pragma unroll
  for (int t = 0; t < 8; ++t) {
    int col = nb + t * 16 + lanelo;
    float bb = bq[col];
#pragma unroll
    for (int i = 0; i < 8; ++i) {
      int r = mb + i + hi8;
      A2[r * 264 + col] = f2bf(acc[t][i] + bb);
    }
  }
  __syncthreads();
  // qc = q @ gW1q^T + gb1
  gemm8(A2, mb, wbf + 262144 /*gW1q*/, nb, acc, lane);
#pragma unroll
  for (int t = 0; t < 8; ++t) {
    int col = nb + t * 16 + lanelo;
    float bb = gb1[col];
#pragma unroll
    for (int i = 0; i < 8; ++i) {
      int r = mb + i + hi8;
      qc[(size_t)(rowBase + r) * DD + col] = acc[t][i] + bb;
    }
  }
}

// ---------------- fused node kernel: gate GEMM + mask + pooled sum ----------------
// One block per graph (256 nodes). 512 threads = 16 waves, one 16-row strip each.
// k-tile and gW1k staged into LDS via CDNA5 async global->LDS copies (ASYNCcnt).
template <bool FINAL>
__global__ void __launch_bounds__(512) k_node_gate(
    const unsigned short* __restrict__ kbf, const unsigned short* __restrict__ wbf,
    const float* __restrict__ qc, const float* __restrict__ gW2, const float* __restrict__ gb2,
    const float* __restrict__ v_ws, float* __restrict__ updates,
    float* __restrict__ out1, float* __restrict__ out2) {
  extern __shared__ __align__(16) unsigned char smem[];
  unsigned short* KT = (unsigned short*)smem;   // [256][264] k tile bf16
  unsigned short* WT = KT + 256 * 264;          // [256][264] gW1k bf16
  float* wrow = (float*)(WT + 256 * 264);       // [256] gate per node
  float* upd  = wrow + 256;                     // [256] pooled accumulator
  const int tid = threadIdx.x, lane = tid & 31, w = tid >> 5;
  const int g = blockIdx.x;
  const unsigned short* gW1k = wbf + 196608;
  // async stage: k tile + weight tile into LDS (bypass VGPRs, ASYNCcnt tracked)
  for (int i = 0; i < 16; ++i) {
    int c = tid + i * 512;
    int row = c >> 5, off = (c & 31) << 3;
    async_ld_b128(lds_off32(KT + row * 264 + off),
                  (const void*)(kbf + (size_t)(g * 256 + row) * DD + off));
    async_ld_b128(lds_off32(WT + row * 264 + off),
                  (const void*)(gW1k + row * DD + off));
  }
  if (tid < 256) upd[tid] = 0.f;
  // prefetch the v rows this wave consumes in the pooling phase
  {
    const float* vb = v_ws + (size_t)(g * 256 + w * 16) * DD + lane * 8;
#pragma unroll
    for (int r = 0; r < 16; ++r) __builtin_prefetch(vb + r * DD, 0, 0);
  }
  wait_asynccnt0();
  __syncthreads();
  // h = relu(K @ gW1k^T + qc[g])  (kept in registers only)
  const int mb = w * 16;
  v8f acc[16];
#pragma unroll
  for (int t = 0; t < 16; ++t) acc[t] = vzero8();
  for (int kk = 0; kk < 256; kk += 32) {
    v16bf a = load_frag(KT, 264, mb, kk, lane);
#pragma unroll
    for (int t = 0; t < 16; ++t) {
      v16bf b = load_frag(WT, 264, t * 16, kk, lane);
      acc[t] = wmma_bf16(a, b, acc[t]);
    }
  }
  const int lanelo = lane & 15, hi8 = (lane >> 4) << 3;
  float p[8];
#pragma unroll
  for (int i = 0; i < 8; ++i) p[i] = 0.f;
#pragma unroll
  for (int t = 0; t < 16; ++t) {
    int col = t * 16 + lanelo;
    float qv = qc[(size_t)g * DD + col];
    float g2 = gW2[col];
#pragma unroll
    for (int i = 0; i < 8; ++i) {
      float h = acc[t][i] + qv;
      h = h > 0.f ? h : 0.f;
      p[i] += h * g2;
    }
  }
  float g2b = gb2[0];
#pragma unroll
  for (int i = 0; i < 8; ++i) {
    float pv = p[i];
    pv += __shfl_xor(pv, 1); pv += __shfl_xor(pv, 2);
    pv += __shfl_xor(pv, 4); pv += __shfl_xor(pv, 8);
    if (lanelo == 0) wrow[mb + i + hi8] = sigmoidf(pv + g2b);
  }
  __syncthreads();
  // mask = sigmoid(k) >= w ; pooled updates ; (final) masked outputs
  float accv[8];
#pragma unroll
  for (int j = 0; j < 8; ++j) accv[j] = 0.f;
  const int d0 = lane * 8;
  for (int r = 0; r < 16; ++r) {
    int m = mb + r;
    float wm = wrow[m];
    const float* vr = v_ws + (size_t)(g * 256 + m) * DD + d0;
    float4 v0 = *(const float4*)vr;
    float4 v1 = *(const float4*)(vr + 4);
    float vv[8] = {v0.x, v0.y, v0.z, v0.w, v1.x, v1.y, v1.z, v1.w};
    float mv[8];
#pragma unroll
    for (int j = 0; j < 8; ++j) {
      float kv = bf2f(KT[m * 264 + d0 + j]);
      bool mk = sigmoidf(kv) >= wm;
      mv[j] = mk ? vv[j] : 0.f;
      accv[j] += mv[j];
    }
    if (FINAL) {
      size_t ob = (size_t)(g * 256 + m) * DD + d0;
      float4 o0 = {mv[0], mv[1], mv[2], mv[3]};
      float4 o1 = {mv[4], mv[5], mv[6], mv[7]};
      float4 q0 = {vv[0] - mv[0], vv[1] - mv[1], vv[2] - mv[2], vv[3] - mv[3]};
      float4 q1 = {vv[4] - mv[4], vv[5] - mv[5], vv[6] - mv[6], vv[7] - mv[7]};
      *(float4*)(out1 + ob) = o0; *(float4*)(out1 + ob + 4) = o1;
      *(float4*)(out2 + ob) = q0; *(float4*)(out2 + ob + 4) = q1;
    }
  }
#pragma unroll
  for (int j = 0; j < 8; ++j) atomicAdd(&upd[d0 + j], accv[j]);
  if (!FINAL) {
    __syncthreads();
    if (tid < 256) updates[(size_t)g * DD + tid] = upd[tid];
  }
}

// ---------------- GRU cell + residual MLP on slots ----------------
__global__ void __launch_bounds__(256) k_gru_mlp(
    float* __restrict__ slots, const float* __restrict__ updates,
    const unsigned short* __restrict__ wbf,
    const float* __restrict__ b_ih, const float* __restrict__ b_hh,
    const float* __restrict__ mb1, const float* __restrict__ mb2,
    const float* __restrict__ lng, const float* __restrict__ lnb) {
  extern __shared__ __align__(16) unsigned char smem[];
  unsigned short* Au = (unsigned short*)smem;   // [64][264] updates bf16, later MLP hidden
  unsigned short* As = Au + 64 * 264;           // [64][264] slots bf16, later LN bf16
  float* Sm = (float*)(As + 64 * 264);          // [64][264] slots f32 (prev -> mid)
  const int tid = threadIdx.x, lane = tid & 31, w = tid >> 5;
  const int rowBase = blockIdx.x * 64;
  for (int i = 0; i < 64; ++i) {
    int e = tid + i * 256;
    int row = e >> 8, col = e & 255;
    float uv = updates[(size_t)(rowBase + row) * DD + col];
    float sv = slots[(size_t)(rowBase + row) * DD + col];
    Au[row * 264 + col] = f2bf(uv);
    As[row * 264 + col] = f2bf(sv);
    Sm[row * 264 + col] = sv;
  }
  __syncthreads();
  const int strip = w & 3, nh = w >> 2;
  const int lanelo = lane & 15, hi8 = (lane >> 4) << 3;
  const int mb = strip * 16, nb = nh * 128;
  const unsigned short* Wih = wbf + 327680;
  const unsigned short* Whh = wbf + 524288;
  v8f t1[8], t2[8], rr[8], zz[8];
  // r gate
  gemm8(Au, mb, Wih, nb, t1, lane);
  gemm8(As, mb, Whh, nb, t2, lane);
#pragma unroll
  for (int t = 0; t < 8; ++t) {
    int col = nb + t * 16 + lanelo;
    float bi = b_ih[col], bh = b_hh[col];
#pragma unroll
    for (int i = 0; i < 8; ++i) rr[t][i] = sigmoidf(t1[t][i] + bi + t2[t][i] + bh);
  }
  // z gate
  gemm8(Au, mb, Wih + 65536, nb, t1, lane);
  gemm8(As, mb, Whh + 65536, nb, t2, lane);
#pragma unroll
  for (int t = 0; t < 8; ++t) {
    int col = nb + t * 16 + lanelo;
    float bi = b_ih[256 + col], bh = b_hh[256 + col];
#pragma unroll
    for (int i = 0; i < 8; ++i) zz[t][i] = sigmoidf(t1[t][i] + bi + t2[t][i] + bh);
  }
  // n candidate + combine
  gemm8(Au, mb, Wih + 131072, nb, t1, lane);
  gemm8(As, mb, Whh + 131072, nb, t2, lane);
#pragma unroll
  for (int t = 0; t < 8; ++t) {
    int col = nb + t * 16 + lanelo;
    float bi = b_ih[512 + col], bh = b_hh[512 + col];
#pragma unroll
    for (int i = 0; i < 8; ++i) {
      int row = mb + i + hi8;
      float nnv = tanhf(t1[t][i] + bi + rr[t][i] * (t2[t][i] + bh));
      float sp = Sm[row * 264 + col];
      Sm[row * 264 + col] = (1.f - zz[t][i]) * nnv + zz[t][i] * sp;
    }
  }
  __syncthreads();
  // residual MLP: LN -> mW1 -> relu -> mW2 -> add
  for (int j = 0; j < 8; ++j) {
    int r = w * 8 + j;
    ln_row(Sm + r * 264, lng, lnb, As + r * 264, nullptr, lane);
  }
  __syncthreads();
  gemm8(As, mb, wbf + 720896 /*mW1*/, nb, t1, lane);
#pragma unroll
  for (int t = 0; t < 8; ++t) {
    int col = nb + t * 16 + lanelo;
    float bb = mb1[col];
#pragma unroll
    for (int i = 0; i < 8; ++i) {
      int row = mb + i + hi8;
      float h = t1[t][i] + bb;
      Au[row * 264 + col] = f2bf(h > 0.f ? h : 0.f);
    }
  }
  __syncthreads();
  gemm8(Au, mb, wbf + 786432 /*mW2*/, nb, t1, lane);
#pragma unroll
  for (int t = 0; t < 8; ++t) {
    int col = nb + t * 16 + lanelo;
    float bb = mb2[col];
#pragma unroll
    for (int i = 0; i < 8; ++i) {
      int row = mb + i + hi8;
      slots[(size_t)(rowBase + row) * DD + col] = Sm[row * 264 + col] + t1[t][i] + bb;
    }
  }
}

// ---------------- host launcher ----------------
extern "C" void kernel_launch(void* const* d_in, const int* in_sizes, int n_in,
                              void* d_out, int out_size, void* d_ws, size_t ws_size,
                              hipStream_t stream) {
  const float* inputs  = (const float*)d_in[0];
  const float* noise   = (const float*)d_in[2];
  const float* mu      = (const float*)d_in[3];
  const float* logsig  = (const float*)d_in[4];
  const float* Wq = (const float*)d_in[5];  const float* bq = (const float*)d_in[6];
  const float* Wk = (const float*)d_in[7];  const float* bk = (const float*)d_in[8];
  const float* Wv = (const float*)d_in[9];  const float* bv = (const float*)d_in[10];
  const float* W_ih = (const float*)d_in[11]; const float* b_ih = (const float*)d_in[12];
  const float* W_hh = (const float*)d_in[13]; const float* b_hh = (const float*)d_in[14];
  const float* gW1 = (const float*)d_in[15]; const float* gb1 = (const float*)d_in[16];
  const float* gW2 = (const float*)d_in[17]; const float* gb2 = (const float*)d_in[18];
  const float* mW1 = (const float*)d_in[19]; const float* mb1 = (const float*)d_in[20];
  const float* mW2 = (const float*)d_in[21]; const float* mb2 = (const float*)d_in[22];
  const float* ln_in_g = (const float*)d_in[23]; const float* ln_in_b = (const float*)d_in[24];
  const float* ln_sl_g = (const float*)d_in[25]; const float* ln_sl_b = (const float*)d_in[26];
  const float* ln_ff_g = (const float*)d_in[27]; const float* ln_ff_b = (const float*)d_in[28];

  unsigned char* ws = (unsigned char*)d_ws;
  float*          v_ws    = (float*)ws;                               // N*D f32 = 256MB
  unsigned short* kbf     = (unsigned short*)(ws + 268435456ull);     // N*D bf16 = 128MB
  unsigned short* wbf     = (unsigned short*)(ws + 402653184ull);     // bf16 weights
  float*          slots   = (float*)(ws + 404357120ull);              // B*D
  float*          qc      = (float*)(ws + 405405696ull);              // B*D
  float*          updates = (float*)(ws + 406454272ull);              // B*D

  float* snOut = (float*)d_out;
  float* out1  = (float*)d_out + (size_t)BB * DD;
  float* out2  = out1 + (size_t)NN * DD;

  const int L1 = 64 * 264 * 2;                       // 33792
  const int L2 = 2 * 64 * 264 * 2;                   // 67584
  const int L3 = 2 * 256 * 264 * 2 + 2 * 256 * 4;    // 272384 (< 320KB WGP LDS)
  const int L4 = 2 * 64 * 264 * 2 + 64 * 264 * 4;    // 135168

  (void)hipFuncSetAttribute(reinterpret_cast<const void*>(k_ln_kv),
                            hipFuncAttributeMaxDynamicSharedMemorySize, L1);
  (void)hipFuncSetAttribute(reinterpret_cast<const void*>(k_slot_q<false>),
                            hipFuncAttributeMaxDynamicSharedMemorySize, L2);
  (void)hipFuncSetAttribute(reinterpret_cast<const void*>(k_slot_q<true>),
                            hipFuncAttributeMaxDynamicSharedMemorySize, L2);
  (void)hipFuncSetAttribute(reinterpret_cast<const void*>(k_node_gate<false>),
                            hipFuncAttributeMaxDynamicSharedMemorySize, L3);
  (void)hipFuncSetAttribute(reinterpret_cast<const void*>(k_node_gate<true>),
                            hipFuncAttributeMaxDynamicSharedMemorySize, L3);
  (void)hipFuncSetAttribute(reinterpret_cast<const void*>(k_gru_mlp),
                            hipFuncAttributeMaxDynamicSharedMemorySize, L4);

  k_conv_weights<<<3328, 256, 0, stream>>>(Wq, Wk, Wv, gW1, W_ih, W_hh, mW1, mW2, wbf);
  k_init_slots<<<1024, 256, 0, stream>>>(mu, logsig, noise, slots);
  k_ln_kv<<<NN / 64, 256, L1, stream>>>(inputs, ln_in_g, ln_in_b, wbf, bk, bv, kbf, v_ws);

  for (int it = 0; it < NITERS; ++it) {
    k_slot_q<false><<<BB / 64, 256, L2, stream>>>(slots, wbf, bq, gb1, ln_sl_g, ln_sl_b, qc, nullptr);
    k_node_gate<false><<<BB, 512, L3, stream>>>(kbf, wbf, qc, gW2, gb2, v_ws, updates, nullptr, nullptr);
    k_gru_mlp<<<BB / 64, 256, L4, stream>>>(slots, updates, wbf, b_ih, b_hh, mb1, mb2, ln_ff_g, ln_ff_b);
  }
  // final: slots_n -> d_out[0:B*D], masked outputs -> rest of d_out
  k_slot_q<true><<<BB / 64, 256, L2, stream>>>(slots, wbf, bq, gb1, ln_sl_g, ln_sl_b, qc, snOut);
  k_node_gate<true><<<BB, 512, L3, stream>>>(kbf, wbf, qc, gW2, gb2, v_ws, updates, out1, out2);
}
